// GroupedQueryAttention_5463198401137
// MI455X (gfx1250) — compile-verified, matrix operand order checked by reading, not run on hardware
//
#include <hip/hip_runtime.h>
#include <hip/hip_bf16.h>
#include <math.h>

// ---------------- problem constants (match reference) ----------------
#define B_   2
#define T_   2048
#define DIN  4096
#define DOUT 4096
#define NH   32
#define NKV  8
#define DH   128
#define GRP  4           // NH / NKV
#define ROPE_BASE 500000.0f

typedef __attribute__((ext_vector_type(16))) __bf16 v16bf;
typedef __attribute__((ext_vector_type(8)))  float  v8f;

__device__ __forceinline__ float bf2f(unsigned short x) {
    return __uint_as_float(((unsigned int)x) << 16);
}
__device__ __forceinline__ unsigned short f2bf(float f) {
    unsigned int u = __float_as_uint(f);
    unsigned int r = u + 0x7FFFu + ((u >> 16) & 1u);   // round-to-nearest-even
    return (unsigned short)(r >> 16);
}
// LDS byte offset of a __shared__ object (what GLOBAL_LOAD_ASYNC_TO_LDS's
// VDST VGPR must hold): addrspace-cast generic -> local, then truncate.
__device__ __forceinline__ unsigned lds_off(const void* p) {
    return (unsigned)(size_t)(__attribute__((address_space(3))) const void*)p;
}
// async copy 64B: global (contiguous) -> LDS (contiguous), per lane
__device__ __forceinline__ void async_copy64(unsigned ldsoff, const void* gptr) {
    asm volatile(
        "global_load_async_to_lds_b128 %0, %1, off\n\t"
        "global_load_async_to_lds_b128 %0, %1, off offset:16\n\t"
        "global_load_async_to_lds_b128 %0, %1, off offset:32\n\t"
        "global_load_async_to_lds_b128 %0, %1, off offset:48"
        :: "v"(ldsoff), "v"(gptr) : "memory");
}
// async copy 32B
__device__ __forceinline__ void async_copy32(unsigned ldsoff, const void* gptr) {
    asm volatile(
        "global_load_async_to_lds_b128 %0, %1, off\n\t"
        "global_load_async_to_lds_b128 %0, %1, off offset:16"
        :: "v"(ldsoff), "v"(gptr) : "memory");
}
__device__ __forceinline__ void wait_async0() {
    asm volatile("s_wait_asynccnt 0x0" ::: "memory");
}

// ---------------- f32 -> bf16 convert ----------------
__global__ void cvt_f32_bf16(const float* __restrict__ src,
                             unsigned short* __restrict__ dst, int n) {
    int i = blockIdx.x * blockDim.x + threadIdx.x;
    int stride = gridDim.x * blockDim.x;
    for (; i < n; i += stride) dst[i] = f2bf(src[i]);
}

// ---------------- tiled bf16 WMMA GEMM: C[M,N] = A[M,K] @ B[K,N] ----------------
// Block tile 128x128, 256 threads = 8 waves (4 in M x 2 in N),
// each wave computes 32x64 via 2x4 wmma_f32_16x16x32_bf16 accumulators.
// A tile staged via GLOBAL_LOAD_ASYNC_TO_LDS (ASYNCcnt); B tile staged
// transposed [col][k] via VGPR scatter so both fragments load as two
// contiguous 16B LDS reads (ISA 16-bit A layout: lanes 0-15 -> K {0..7,16..23},
// lanes 16-31 -> K {8..15,24..31}).
template <bool OUT_F32>
__global__ __launch_bounds__(256)
void gemm_bf16(const unsigned short* __restrict__ A,
               const unsigned short* __restrict__ Bm,
               void* __restrict__ Cp, int M, int N, int K) {
    __shared__ unsigned short As[128][40];    // stride pad vs bank conflicts
    __shared__ unsigned short Bst[128][40];   // B transposed: [col][k]

    const int tid  = threadIdx.x;
    const int wid  = tid >> 5;
    const int lane = tid & 31;
    const int wm   = wid >> 1;      // 0..3
    const int wn   = wid & 1;       // 0..1
    const int m0   = blockIdx.y * 128;
    const int n0   = blockIdx.x * 128;

    v8f acc[2][4];
#pragma unroll
    for (int i = 0; i < 2; i++)
#pragma unroll
        for (int j = 0; j < 4; j++)
            acc[i][j] = (v8f){0.f,0.f,0.f,0.f,0.f,0.f,0.f,0.f};

    const int ar = tid >> 1, ah = (tid & 1) * 16;   // A loader: row, k-half
    const int br = tid >> 3, bs = (tid & 7) * 16;   // B loader: k-row, col-seg
    const int kb = (lane >= 16) ? 8 : 0;
    const int rl = lane & 15;
    const unsigned aoff = lds_off(&As[ar][ah]);

    for (int k0 = 0; k0 < K; k0 += 32) {
        __syncthreads();
        // stage A tile (128x32): async DMA global -> LDS, 32B per thread
        const unsigned short* ap = A + (size_t)(m0 + ar) * K + k0 + ah;
        __builtin_prefetch(ap + 32, 0, 0);             // global_prefetch next k-tile
        async_copy32(aoff, ap);
        // stage B tile (32x128) transposed into Bst[col][k]
        const unsigned short* bp = Bm + (size_t)(k0 + br) * N + n0 + bs;
        union { uint4 u[2]; unsigned short s[16]; } bld;
        bld.u[0] = *(const uint4*)bp;
        bld.u[1] = *(const uint4*)(bp + 8);
#pragma unroll
        for (int c = 0; c < 16; c++) Bst[bs + c][br] = bld.s[c];
        wait_async0();
        __syncthreads();

#pragma unroll
        for (int mi = 0; mi < 2; mi++) {
            union { v16bf v; uint4 u[2]; } af;
            const unsigned short* arow = &As[wm * 32 + mi * 16 + rl][0];
            af.u[0] = *(const uint4*)(arow + kb);
            af.u[1] = *(const uint4*)(arow + kb + 16);
#pragma unroll
            for (int ni = 0; ni < 4; ni++) {
                union { v16bf v; uint4 u[2]; } bfm;
                const unsigned short* brow = &Bst[wn * 64 + ni * 16 + rl][0];
                bfm.u[0] = *(const uint4*)(brow + kb);
                bfm.u[1] = *(const uint4*)(brow + kb + 16);
                acc[mi][ni] = __builtin_amdgcn_wmma_f32_16x16x32_bf16(
                    false, af.v, false, bfm.v, (short)0, acc[mi][ni], false, false);
            }
        }
    }

    // epilogue: C layout -> row = base + (lane>=16 ? 8:0) + vgpr_r, col = base + lane%16
    const int roff = (lane >= 16) ? 8 : 0;
#pragma unroll
    for (int mi = 0; mi < 2; mi++)
#pragma unroll
        for (int ni = 0; ni < 4; ni++) {
            int row = m0 + wm * 32 + mi * 16 + roff;
            int col = n0 + wn * 64 + ni * 16 + rl;
#pragma unroll
            for (int r = 0; r < 8; r++) {
                float v = acc[mi][ni][r];
                if (OUT_F32)
                    ((float*)Cp)[(size_t)(row + r) * N + col] = v;
                else
                    ((unsigned short*)Cp)[(size_t)(row + r) * N + col] = f2bf(v);
            }
        }
}

// ---------------- RoPE + head permute ----------------
// src: (B, T, nheads*DH) bf16 (projection output)
// dst: (B, nheads, T, DH) bf16, rotated
__global__ void rope_permute(const unsigned short* __restrict__ src,
                             unsigned short* __restrict__ dst, int nheads) {
    int idx = blockIdx.x * blockDim.x + threadIdx.x;
    int total = B_ * T_ * nheads * (DH / 2);
    if (idx >= total) return;
    int i   = idx % (DH / 2);
    int tmp = idx / (DH / 2);
    int h   = tmp % nheads;  tmp /= nheads;
    int t   = tmp % T_;
    int b   = tmp / T_;

    float inv = __expf(-((2.0f * (float)i) / (float)DH) * __logf(ROPE_BASE));
    float ang = (float)t * inv;
    float s, c;
    __sincosf(ang, &s, &c);

    size_t so = ((size_t)(b * T_ + t)) * ((size_t)nheads * DH) + (size_t)h * DH + i;
    float x1 = bf2f(src[so]);
    float x2 = bf2f(src[so + DH / 2]);
    size_t dofs = (((size_t)(b * nheads + h)) * T_ + t) * DH + i;
    dst[dofs]          = f2bf(x1 * c - x2 * s);
    dst[dofs + DH / 2] = f2bf(x1 * s + x2 * c);
}

// ---------------- causal GQA flash attention ----------------
// Q: (B,NH,T,DH) bf16 (roped), Kr: (B,NKV,T,DH) bf16 (roped),
// Vp: (B,T,NKV,DH) bf16 (raw proj), ctx out: (B,T,NH*DH) bf16.
// Block = 128 threads (4 waves), 64 query rows/block, 16 rows/wave.
// K tile staged via async DMA to LDS; V tile transposed via VGPR scatter.
__global__ __launch_bounds__(128)
void attn_gqa(const unsigned short* __restrict__ Q,
              const unsigned short* __restrict__ Kr,
              const unsigned short* __restrict__ Vp,
              unsigned short* __restrict__ ctx) {
    __shared__ unsigned short Ks[32][136];    // K tile row-major [key][d]
    __shared__ unsigned short Vst[128][40];   // V tile transposed [d][key]
    __shared__ unsigned short Ps[4][16][40];  // per-wave P scratch [row][key]

    const int tid  = threadIdx.x;
    const int wid  = tid >> 5;
    const int lane = tid & 31;
    const int q0   = blockIdx.x * 64;
    const int h    = blockIdx.y;
    const int b    = blockIdx.z;
    const int g    = h >> 2;                  // GQA group
    const int kb   = (lane >= 16) ? 8 : 0;
    const int rl   = lane & 15;
    const int roff = (lane >= 16) ? 8 : 0;

    union F { v16bf v; uint4 u[2]; };

    // load this wave's Q block (16 x 128) as 4 A-fragments, kept in registers
    F qf[4];
    const int qrowl = q0 + wid * 16 + rl;
    const unsigned short* qbase =
        Q + (((size_t)(b * NH + h)) * T_ + qrowl) * DH;
#pragma unroll
    for (int c = 0; c < 4; c++) {
        qf[c].u[0] = *(const uint4*)(qbase + c * 32 + kb);
        qf[c].u[1] = *(const uint4*)(qbase + c * 32 + kb + 16);
    }

    float mrow[8], lrow[8];
    v8f acc[8];
#pragma unroll
    for (int r = 0; r < 8; r++) { mrow[r] = -3.0e38f; lrow[r] = 0.f; }
#pragma unroll
    for (int d = 0; d < 8; d++) acc[d] = (v8f){0.f,0.f,0.f,0.f,0.f,0.f,0.f,0.f};

    const float scale = 0.08838834764831845f;  // 1/sqrt(128)

    const int key = tid >> 2, seg = (tid & 3) * 32;   // staging roles
    const unsigned ksoff = lds_off(&Ks[key][seg]);

    for (int kt = 0; kt <= q0 + 63; kt += 32) {
        __syncthreads();
        // cooperative stage: K tile (32x128) via async DMA (64B per thread)
        const unsigned short* kp =
            Kr + (((size_t)(b * NKV + g)) * T_ + kt + key) * DH + seg;
        async_copy64(ksoff, kp);
        // V tile transposed via VGPR scatter
        {
            const unsigned short* vp =
                Vp + (((size_t)(b * T_ + kt + key)) * NKV + g) * DH + seg;
            union { uint4 u[4]; unsigned short s[32]; } vld;
            vld.u[0] = ((const uint4*)vp)[0];
            vld.u[1] = ((const uint4*)vp)[1];
            vld.u[2] = ((const uint4*)vp)[2];
            vld.u[3] = ((const uint4*)vp)[3];
#pragma unroll
            for (int c = 0; c < 32; c++) Vst[seg + c][key] = vld.s[c];
        }
        wait_async0();
        __syncthreads();

        // S = Q K^T : two 16x16 key blocks, K-dim 128 = 4 chunks of 32
        v8f s[2];
#pragma unroll
        for (int j = 0; j < 2; j++) {
            s[j] = (v8f){0.f,0.f,0.f,0.f,0.f,0.f,0.f,0.f};
#pragma unroll
            for (int c = 0; c < 4; c++) {
                F kf;
                const unsigned short* krow = &Ks[j * 16 + rl][c * 32];
                kf.u[0] = *(const uint4*)(krow + kb);
                kf.u[1] = *(const uint4*)(krow + kb + 16);
                s[j] = __builtin_amdgcn_wmma_f32_16x16x32_bf16(
                    false, qf[c].v, false, kf.v, (short)0, s[j], false, false);
            }
        }

        // scale + causal mask + online softmax (row spread over 16 lanes)
        float p0[8], p1[8];
#pragma unroll
        for (int r = 0; r < 8; r++) {
            int qrow = q0 + wid * 16 + roff + r;
            float v0 = s[0][r] * scale;
            float v1 = s[1][r] * scale;
            if (kt + rl      > qrow) v0 = -3.0e38f;
            if (kt + 16 + rl > qrow) v1 = -3.0e38f;
            float mx = fmaxf(v0, v1);
            mx = fmaxf(mx, __shfl_xor(mx, 1));
            mx = fmaxf(mx, __shfl_xor(mx, 2));
            mx = fmaxf(mx, __shfl_xor(mx, 4));
            mx = fmaxf(mx, __shfl_xor(mx, 8));
            float mnew  = fmaxf(mrow[r], mx);
            float alpha = __expf(mrow[r] - mnew);
            float e0 = __expf(v0 - mnew);
            float e1 = __expf(v1 - mnew);
            float ps = e0 + e1;
            ps += __shfl_xor(ps, 1);
            ps += __shfl_xor(ps, 2);
            ps += __shfl_xor(ps, 4);
            ps += __shfl_xor(ps, 8);
            lrow[r] = lrow[r] * alpha + ps;
            mrow[r] = mnew;
            p0[r] = e0; p1[r] = e1;
#pragma unroll
            for (int d = 0; d < 8; d++) acc[d][r] *= alpha;
        }

        // C-layout P -> LDS -> A-fragment (transpose through per-wave scratch)
#pragma unroll
        for (int r = 0; r < 8; r++) {
            Ps[wid][roff + r][rl]      = f2bf(p0[r]);
            Ps[wid][roff + r][16 + rl] = f2bf(p1[r]);
        }
        __syncthreads();

        F pf;
        const unsigned short* prow = &Ps[wid][rl][0];
        pf.u[0] = *(const uint4*)(prow + kb);
        pf.u[1] = *(const uint4*)(prow + kb + 16);
#pragma unroll
        for (int d = 0; d < 8; d++) {
            F vf;
            const unsigned short* vrow = &Vst[d * 16 + rl][0];
            vf.u[0] = *(const uint4*)(vrow + kb);
            vf.u[1] = *(const uint4*)(vrow + kb + 16);
            acc[d] = __builtin_amdgcn_wmma_f32_16x16x32_bf16(
                false, pf.v, false, vf.v, (short)0, acc[d], false, false);
        }
    }

    // normalize + write ctx (B,T,NH*DH) bf16
#pragma unroll
    for (int d = 0; d < 8; d++) {
#pragma unroll
        for (int r = 0; r < 8; r++) {
            int qrow = q0 + wid * 16 + roff + r;
            float v = acc[d][r] / lrow[r];
            size_t o = ((size_t)(b * T_ + qrow)) * DOUT + (size_t)h * DH + d * 16 + rl;
            ctx[o] = f2bf(v);
        }
    }
}

// ---------------- host-side orchestration ----------------
extern "C" void kernel_launch(void* const* d_in, const int* in_sizes, int n_in,
                              void* d_out, int out_size, void* d_ws, size_t ws_size,
                              hipStream_t stream) {
    (void)in_sizes; (void)n_in; (void)out_size; (void)ws_size;
    const float* x  = (const float*)d_in[0];
    const float* Wq = (const float*)d_in[1];
    const float* Wk = (const float*)d_in[2];
    const float* Wv = (const float*)d_in[3];
    const float* Wo = (const float*)d_in[4];
    float* out = (float*)d_out;

    char* ws = (char*)d_ws;
    const size_t MB = 1024ull * 1024ull;
    unsigned short* xb  = (unsigned short*)(ws + 0 * MB);    // 32 MB
    unsigned short* Wqb = (unsigned short*)(ws + 32 * MB);   // 32 MB
    unsigned short* Wkb = (unsigned short*)(ws + 64 * MB);   // 8 MB
    unsigned short* Wvb = (unsigned short*)(ws + 72 * MB);   // 8 MB
    unsigned short* Wob = (unsigned short*)(ws + 80 * MB);   // 32 MB
    unsigned short* Qp  = (unsigned short*)(ws + 112 * MB);  // 32 MB (reused as ctx)
    unsigned short* Kp  = (unsigned short*)(ws + 144 * MB);  // 8 MB
    unsigned short* Vp  = (unsigned short*)(ws + 152 * MB);  // 8 MB
    unsigned short* Qr  = (unsigned short*)(ws + 160 * MB);  // 32 MB
    unsigned short* Krr = (unsigned short*)(ws + 192 * MB);  // 8 MB -> 200 MB total
    unsigned short* ctx = Qp;                                // Qp dead after RoPE

    const int M = B_ * T_;  // 4096

    // 1) f32 -> bf16 converts
    cvt_f32_bf16<<<2048, 256, 0, stream>>>(x,  xb,  M * DIN);
    cvt_f32_bf16<<<2048, 256, 0, stream>>>(Wq, Wqb, DIN * DOUT);
    cvt_f32_bf16<<<2048, 256, 0, stream>>>(Wk, Wkb, DIN * NKV * DH);
    cvt_f32_bf16<<<2048, 256, 0, stream>>>(Wv, Wvb, DIN * NKV * DH);
    cvt_f32_bf16<<<2048, 256, 0, stream>>>(Wo, Wob, DOUT * DOUT);

    // 2) projections (bf16 out)
    gemm_bf16<false><<<dim3(DOUT / 128, M / 128), 256, 0, stream>>>(
        xb, Wqb, (void*)Qp, M, DOUT, DIN);
    gemm_bf16<false><<<dim3((NKV * DH) / 128, M / 128), 256, 0, stream>>>(
        xb, Wkb, (void*)Kp, M, NKV * DH, DIN);
    gemm_bf16<false><<<dim3((NKV * DH) / 128, M / 128), 256, 0, stream>>>(
        xb, Wvb, (void*)Vp, M, NKV * DH, DIN);

    // 3) RoPE + permute to head-major
    {
        int nq = B_ * T_ * NH * (DH / 2);
        rope_permute<<<(nq + 255) / 256, 256, 0, stream>>>(Qp, Qr, NH);
        int nk = B_ * T_ * NKV * (DH / 2);
        rope_permute<<<(nk + 255) / 256, 256, 0, stream>>>(Kp, Krr, NKV);
    }

    // 4) causal GQA flash attention -> ctx (aliases Qp, which is now dead)
    attn_gqa<<<dim3(T_ / 64, NH, B_), 128, 0, stream>>>(Qr, Krr, Vp, ctx);

    // 5) output projection (f32 out)
    gemm_bf16<true><<<dim3(DOUT / 128, M / 128), 256, 0, stream>>>(
        ctx, Wob, (void*)out, M, DOUT, DOUT);
}
